// KmeansClusteringLayer_65798898975201
// MI455X (gfx1250) — matrix operand order repaired and checked by reading, not compiled, not run on hardware
//
#include <hip/hip_runtime.h>
#include <stdint.h>

#define N_PTS 131072
#define DIM   512
#define KC    1024

typedef __bf16 v16bf __attribute__((ext_vector_type(16)));
typedef float  v8f   __attribute__((ext_vector_type(8)));
typedef float  nf4   __attribute__((ext_vector_type(4)));   // native float4 (real vector type)
typedef int    v4i   __attribute__((vector_size(16)));       // matches async-builtin param type
typedef __attribute__((address_space(3))) v4i* lds_v4i_p;

union FragBF { v16bf v; unsigned short u[16]; };

__device__ __forceinline__ unsigned short f2bf(float f) {
    // round-to-nearest-even f32 -> bf16 (bit trick; NaN edge irrelevant here)
    unsigned int u = __float_as_uint(f);
    u += 0x7FFFu + ((u >> 16) & 1u);
    return (unsigned short)(u >> 16);
}

__device__ __forceinline__ void wait_async_copies() {
#if __has_builtin(__builtin_amdgcn_s_wait_asynccnt)
    __builtin_amdgcn_s_wait_asynccnt(0);
#else
    asm volatile("s_wait_asynccnt 0x0" ::: "memory");
#endif
}

// ---- prep: centroids [D][K] f32 -> centroidsT [K][D] bf16 (transpose) ----
__global__ void km_prep_transpose(const float* __restrict__ cen,
                                  unsigned short* __restrict__ centT) {
    __shared__ float tile[16][17];
    const int k0 = blockIdx.x * 16, d0 = blockIdx.y * 16;
    const int tx = threadIdx.x, ty = threadIdx.y;
    tile[ty][tx] = cen[(size_t)(d0 + ty) * KC + (k0 + tx)];
    __syncthreads();
    centT[(size_t)(k0 + ty) * DIM + (d0 + tx)] = f2bf(tile[tx][ty]);
}

// ---- prep: c_sq[k] = sum_d centroids[d][k]^2 (coalesced across k) ----
__global__ void km_prep_csq(const float* __restrict__ cen,
                            float* __restrict__ c_sq) {
    const int k = blockIdx.x * 256 + threadIdx.x;
    float s = 0.0f;
    for (int d = 0; d < DIM; ++d) {
        float v = cen[(size_t)d * KC + k];
        s += v * v;
    }
    c_sq[k] = s;
}

// ---- main: WMMA cross-term GEMM + fused argmin, double-buffered B staging ----
__launch_bounds__(256)
__global__ void km_assign(const float* __restrict__ feat,
                          const unsigned short* __restrict__ centT,
                          const float* __restrict__ c_sq,
                          float* __restrict__ out) {
    __shared__ unsigned short Btile[2][16 * DIM];   // 2 x 16 KB double buffer

    const int tid     = threadIdx.x;
    const int lane    = tid & 31;
    const int wave    = tid >> 5;
    const int rowbase = blockIdx.x * 128 + wave * 16;

    // cooperative stage of k-tile `kt` into LDS buffer `buf` (issue only, no wait)
    auto stage = [&](int kt, int buf) {
        unsigned short* gsrc =
            const_cast<unsigned short*>(centT) + (size_t)kt * 16 * DIM;
#if __has_builtin(__builtin_amdgcn_global_load_async_to_lds_b128)
        #pragma unroll
        for (int j = 0; j < 4; ++j) {
            const int e = (tid + j * 256) * 8;           // 8 ushorts = 16 bytes
            __builtin_amdgcn_global_load_async_to_lds_b128(
                (v4i*)(gsrc + e),
                (lds_v4i_p)&Btile[buf][e],
                /*offset=*/0, /*cpol=*/0);
        }
#else
        const uint4* gs = (const uint4*)gsrc;
        uint4* ls = (uint4*)Btile[buf];
        #pragma unroll
        for (int j = 0; j < 4; ++j) ls[tid + j * 256] = gs[tid + j * 256];
#endif
    };

    // ---- A panel: 16 rows x 512 d, converted f32->bf16, held in registers ----
    // A 16x32 bf16 layout: lanes 0-15 = row lane, hold K 0..15 contiguous;
    //                      lanes 16-31 = row lane-16, hold K 16..31.
    stage(0, 0);   // overlap first B staging with the A-panel conversion
    FragBF a[16];
    {
        const float* arow = feat + (size_t)(rowbase + (lane & 15)) * DIM;
        const int off0 = (lane < 16) ? 0 : 16;
        #pragma unroll
        for (int dc = 0; dc < 16; ++dc) {
            const int b0 = dc * 32 + off0;
            nf4 f0 = __builtin_nontemporal_load((const nf4*)(arow + b0));
            nf4 f1 = __builtin_nontemporal_load((const nf4*)(arow + b0 + 4));
            nf4 f2 = __builtin_nontemporal_load((const nf4*)(arow + b0 + 8));
            nf4 f3 = __builtin_nontemporal_load((const nf4*)(arow + b0 + 12));
            float f[16] = { f0.x, f0.y, f0.z, f0.w,  f1.x, f1.y, f1.z, f1.w,
                            f2.x, f2.y, f2.z, f2.w,  f3.x, f3.y, f3.z, f3.w };
            #pragma unroll
            for (int j = 0; j < 16; ++j) a[dc].u[j] = f2bf(f[j]);
        }
    }

    float minv[8];
    int   mini[8];
    #pragma unroll
    for (int i = 0; i < 8; ++i) { minv[i] = 3.4e38f; mini[i] = 0; }

    // B 32x16 bf16 layout: lanes 0-15 = col lane, K 0..15 contiguous;
    //                      lanes 16-31 = col lane-16, K 16..31.
    const int col   = lane & 15;
    const int dhalf = (lane < 16) ? 0 : 16;

    for (int kt = 0; kt < 64; ++kt) {
        const int b = kt & 1;
        // finish this wave's staging of tile kt, then sync so ALL waves'
        // pieces are visible; the barrier also proves every wave is done
        // computing on buffer 1-b, so it is safe to restage it.
        wait_async_copies();
        __syncthreads();
        if (kt + 1 < 64) stage(kt + 1, 1 - b);   // overlaps the WMMA burst below

        v8f acc = {};
        #pragma unroll
        for (int dc = 0; dc < 16; ++dc) {
            FragBF bf;
            bf.v = *reinterpret_cast<const v16bf*>(&Btile[b][col * DIM + dc * 32 + dhalf]);
            acc = __builtin_amdgcn_wmma_f32_16x16x32_bf16(
                      /*neg_a=*/false, a[dc].v, /*neg_b=*/false, bf.v,
                      /*c_mod=*/(short)0, acc, /*reuse_a=*/false, /*reuse_b=*/false);
        }

        // score(n, m) = c_sq[k] - 2 * cross  (x_sq is constant per row -> dropped)
        const float csq = c_sq[kt * 16 + col];
        const int   kk  = kt * 16 + col;
        #pragma unroll
        for (int i = 0; i < 8; ++i) {
            float score = csq - 2.0f * acc[i];
            if (score < minv[i]) { minv[i] = score; mini[i] = kk; }
        }
    }

    // cross-lane argmin within each 16-lane half (C/D layout: lanes 0-15 rows 0-7,
    // lanes 16-31 rows 8-15; xor masks 1/2/4/8 never cross the halves)
    #pragma unroll
    for (int m = 1; m < 16; m <<= 1) {
        #pragma unroll
        for (int i = 0; i < 8; ++i) {
            float ov = __shfl_xor(minv[i], m);
            int   oi = __shfl_xor(mini[i], m);
            if (ov < minv[i] || (ov == minv[i] && oi < mini[i])) {
                minv[i] = ov; mini[i] = oi;
            }
        }
    }
    if (col == 0) {
        const int half = (lane < 16) ? 0 : 8;
        #pragma unroll
        for (int i = 0; i < 8; ++i)
            out[rowbase + half + i] = (float)mini[i];
    }
}

extern "C" void kernel_launch(void* const* d_in, const int* in_sizes, int n_in,
                              void* d_out, int out_size, void* d_ws, size_t ws_size,
                              hipStream_t stream) {
    (void)in_sizes; (void)n_in; (void)out_size; (void)ws_size;
    const float* feat = (const float*)d_in[0];   // [N, D] f32
    const float* cen  = (const float*)d_in[1];   // [D, K] f32
    float* out = (float*)d_out;                  // [N] f32 (assignment indices)

    unsigned short* centT = (unsigned short*)d_ws;                       // [K][D] bf16, 1 MB
    float* c_sq = (float*)((char*)d_ws + (size_t)KC * DIM * sizeof(unsigned short)); // 4 KB

    km_prep_transpose<<<dim3(KC / 16, DIM / 16), dim3(16, 16), 0, stream>>>(cen, centT);
    km_prep_csq<<<KC / 256, 256, 0, stream>>>(cen, c_sq);
    km_assign<<<N_PTS / 128, 256, 0, stream>>>(feat, centT, c_sq, out);
}
